// DGCN_120259084553
// MI455X (gfx1250) — matrix operand order repaired and checked by reading, not compiled, not run on hardware
//
#include <hip/hip_runtime.h>
#include <math.h>

// Problem constants (match reference setup)
#define BSZ    16          // batch
#define NN     4096        // nodes
#define INDIM  64
#define HID    128
#define CC     192         // IN_DIM + HID
#define MM     5           // num diffusion matrices
#define DEG    16
#define CB     (CC * BSZ)  // 3072, width of x0/x1 rows
#define KK     (CC * MM)   // 960, GEMM reduction dim
#define ROWS   (BSZ * NN)  // 65536, GEMM M dim

typedef __bf16 v8bf  __attribute__((ext_vector_type(8)));
typedef __bf16 v16bf __attribute__((ext_vector_type(16)));
typedef float  v8f   __attribute__((ext_vector_type(8)));

__device__ __forceinline__ unsigned short f32_to_bf16(float f) {
    unsigned int u = __float_as_uint(f);
    u += 0x7fffu + ((u >> 16) & 1u);   // round-to-nearest-even
    return (unsigned short)(u >> 16);
}

// Branch-free tanh: tanh(x) = copysign(1 - 2*rcp(exp(2|x|)+1), x).
// v_exp_f32 + v_rcp_f32; no exec-mask divergence, no IEEE div sequence;
// saturates to +/-1 when exp overflows to inf (rcp(inf) = 0).
__device__ __forceinline__ float fast_tanh(float x) {
    float ax = fabsf(x);
    float e  = __expf(2.0f * ax);
    float t  = 1.0f - 2.0f * __builtin_amdgcn_rcpf(e + 1.0f);
    return copysignf(t, x);
}

// ---------------------------------------------------------------------------
// Stage 0a: build x0 [N, C*B] fp32 and write m=0 slab of X (bf16, [B*N, 960])
// X index: row r = b*NN + n, col k = c*MM + m
// ---------------------------------------------------------------------------
__global__ void __launch_bounds__(256) build_x0_kernel(
    const float* __restrict__ inputs,   // [B, N, 64]
    const float* __restrict__ state,    // [B, N, 128]
    float* __restrict__ x0,             // [N, 3072]
    unsigned short* __restrict__ X)     // [65536, 960] bf16
{
    int idx = blockIdx.x * blockDim.x + threadIdx.x;
    if (idx >= NN * CB) return;
    int n = idx / CB;
    int j = idx - n * CB;        // j = c*B + b
    int b = j & (BSZ - 1);
    int c = j >> 4;
    float v;
    if (c < INDIM) v = inputs[((size_t)b * NN + n) * INDIM + c];
    else           v = state [((size_t)b * NN + n) * HID + (c - INDIM)];
    x0[idx] = v;
    X[(size_t)(b * NN + n) * KK + c * MM + 0] = f32_to_bf16(v);
}

// ---------------------------------------------------------------------------
// Stage 0b: transpose weights [960,128] fp32 -> WT [128,960] bf16
// ---------------------------------------------------------------------------
__global__ void __launch_bounds__(256) prep_w_kernel(
    const float* __restrict__ W, unsigned short* __restrict__ WT)
{
    int idx = blockIdx.x * blockDim.x + threadIdx.x;
    if (idx >= KK * HID) return;
    int h = idx / KK;
    int k = idx - h * KK;
    WT[idx] = f32_to_bf16(W[(size_t)k * HID + h]);
}

// ---------------------------------------------------------------------------
// Stage 1: x1 = A @ x0. blockIdx.y = n (row) -> edge-list reads are wave-
// uniform and lower to scalar loads; 3072-wide j dimension = 12 x 256 blocks.
// ---------------------------------------------------------------------------
__global__ void __launch_bounds__(256) spmm1_kernel(
    const int*   __restrict__ cols,   // [NNZ] for this support
    const float* __restrict__ vals,   // [NNZ]
    const float* __restrict__ x0,     // [N, 3072]
    float*       __restrict__ x1,     // [N, 3072]
    unsigned short* __restrict__ X, int m)
{
    const int n = blockIdx.y;
    const int j = blockIdx.x * blockDim.x + threadIdx.x;   // 0..3071
    const int*   ce = cols + n * DEG;
    const float* ve = vals + n * DEG;
    float acc = 0.0f;
    #pragma unroll
    for (int d = 0; d < DEG; ++d)
        acc += ve[d] * x0[(size_t)ce[d] * CB + j];
    x1[(size_t)n * CB + j] = acc;
    int b = j & (BSZ - 1);
    int c = j >> 4;
    X[(size_t)(b * NN + n) * KK + c * MM + m] = f32_to_bf16(acc);
}

// ---------------------------------------------------------------------------
// Stage 2: x2 = 2*(A @ x1) - x0 -> only the bf16 slab m of X is needed
// ---------------------------------------------------------------------------
__global__ void __launch_bounds__(256) spmm2_kernel(
    const int*   __restrict__ cols,
    const float* __restrict__ vals,
    const float* __restrict__ x1,
    const float* __restrict__ x0,
    unsigned short* __restrict__ X, int m)
{
    const int n = blockIdx.y;
    const int j = blockIdx.x * blockDim.x + threadIdx.x;
    const int*   ce = cols + n * DEG;
    const float* ve = vals + n * DEG;
    float acc = 0.0f;
    #pragma unroll
    for (int d = 0; d < DEG; ++d)
        acc += ve[d] * x1[(size_t)ce[d] * CB + j];
    float v = 2.0f * acc - x0[(size_t)n * CB + j];
    int b = j & (BSZ - 1);
    int c = j >> 4;
    X[(size_t)(b * NN + n) * KK + c * MM + m] = f32_to_bf16(v);
}

// ---------------------------------------------------------------------------
// Stage 3: out = tanh(X @ WT^T + bias), WMMA bf16 -> fp32.
// Block = 256 thr = 8 waves; block owns 64 rows x 128 cols.
// Wave (wr=w>>2, wc=w&3) owns a 2x2 grid of 16x16 tiles: rows r0/r0+16,
// cols 32wc/32wc+16 -> each A/B fragment feeds two WMMAs (2 loads per wmma).
//
// A frag (16x32 bf16): lane L holds row base+(L&15); khalf=L>>4;
//   elems 0..7  = X[row, k0 + 8*khalf .. +7]
//   elems 8..15 = X[row, k0 + 16 + 8*khalf .. +7]
// B frag (32x16 bf16): lane L holds col base+(L&15);
//   elems 0..7  = WT[col, k0 + 8*khalf .. +7]
//   elems 8..15 = WT[col, k0 + 16 + 8*khalf .. +7]
// C/D frag: VGPR i -> (M = i + 8*khalf, N = lane&15)
// ---------------------------------------------------------------------------
__device__ __forceinline__ v16bf load_a_frag(const unsigned short* p) {
    v8bf lo = *(const v8bf*)(p);
    v8bf hi = *(const v8bf*)(p + 16);
    v16bf a;
    #pragma unroll
    for (int i = 0; i < 8; ++i) { a[i] = lo[i]; a[i + 8] = hi[i]; }
    return a;
}

__device__ __forceinline__ v16bf load_b_frag(const unsigned short* p) {
    v8bf lo = *(const v8bf*)(p);
    v8bf hi = *(const v8bf*)(p + 16);
    v16bf b;
    #pragma unroll
    for (int i = 0; i < 8; ++i) { b[i] = lo[i]; b[i + 8] = hi[i]; }
    return b;
}

__global__ void __launch_bounds__(256) gemm_tanh_kernel(
    const unsigned short* __restrict__ X,    // [65536, 960] bf16
    const unsigned short* __restrict__ WT,   // [128, 960]  bf16
    const float* __restrict__ bias,          // [128]
    float* __restrict__ out)                 // [65536, 128] fp32
{
    const int lane  = threadIdx.x & 31;
    const int wave  = threadIdx.x >> 5;
    const int l15   = lane & 15;
    const int khalf = lane >> 4;
    const int wr    = wave >> 2;             // 0..1
    const int wc    = wave & 3;              // 0..3
    const int r0    = blockIdx.x * 64 + wr * 32;
    const int r1    = r0 + 16;
    const int c0    = wc * 32;
    const int c1    = c0 + 16;

    const unsigned short* a0p = X  + (size_t)(r0 + l15) * KK + khalf * 8;
    const unsigned short* a1p = X  + (size_t)(r1 + l15) * KK + khalf * 8;
    const unsigned short* b0p = WT + (size_t)(c0 + l15) * KK + khalf * 8;
    const unsigned short* b1p = WT + (size_t)(c1 + l15) * KK + khalf * 8;

    v8f acc00 = {}, acc01 = {}, acc10 = {}, acc11 = {};
    #pragma unroll 2
    for (int kb = 0; kb < 30; ++kb) {
        v16bf a0 = load_a_frag(a0p);
        v16bf a1 = load_a_frag(a1p);
        v16bf b0 = load_b_frag(b0p);
        v16bf b1 = load_b_frag(b1p);
        acc00 = __builtin_amdgcn_wmma_f32_16x16x32_bf16(false, a0, false, b0,
                                                        (short)0, acc00, false, false);
        acc01 = __builtin_amdgcn_wmma_f32_16x16x32_bf16(false, a0, false, b1,
                                                        (short)0, acc01, false, false);
        acc10 = __builtin_amdgcn_wmma_f32_16x16x32_bf16(false, a1, false, b0,
                                                        (short)0, acc10, false, false);
        acc11 = __builtin_amdgcn_wmma_f32_16x16x32_bf16(false, a1, false, b1,
                                                        (short)0, acc11, false, false);
        a0p += 32; a1p += 32; b0p += 32; b1p += 32;
    }

    const float bc0 = bias[c0 + l15];
    const float bc1 = bias[c1 + l15];
    #pragma unroll
    for (int i = 0; i < 8; ++i) {
        int ra = r0 + khalf * 8 + i;
        int rb = r1 + khalf * 8 + i;
        out[(size_t)ra * HID + c0 + l15] = fast_tanh(acc00[i] + bc0);
        out[(size_t)ra * HID + c1 + l15] = fast_tanh(acc01[i] + bc1);
        out[(size_t)rb * HID + c0 + l15] = fast_tanh(acc10[i] + bc0);
        out[(size_t)rb * HID + c1 + l15] = fast_tanh(acc11[i] + bc1);
    }
}

// ---------------------------------------------------------------------------
extern "C" void kernel_launch(void* const* d_in, const int* in_sizes, int n_in,
                              void* d_out, int out_size, void* d_ws, size_t ws_size,
                              hipStream_t stream) {
    const float* inputs  = (const float*)d_in[0];   // [16,4096,64]
    const float* state_t = (const float*)d_in[1];   // [16,4096,128]
    const float* weights = (const float*)d_in[2];   // [960,128]
    const float* biases  = (const float*)d_in[3];   // [128]
    // d_in[4] = sup_rows: structured as repeat(arange(N), DEG) -> implicit
    const int*   sup_cols = (const int*)d_in[5];    // [2, 65536]
    const float* sup_vals = (const float*)d_in[6];  // [2, 65536]
    float* out = (float*)d_out;

    // Workspace carve-up
    float* x0 = (float*)d_ws;                                     // 4096*3072 f32
    float* x1 = x0 + (size_t)NN * CB;                             // 4096*3072 f32
    unsigned short* X  = (unsigned short*)(x1 + (size_t)NN * CB); // 65536*960 bf16
    unsigned short* WT = X + (size_t)ROWS * KK;                   // 128*960 bf16

    const int T = 256;
    const int g_feat = (NN * CB + T - 1) / T;       // 49152 blocks
    const int g_w    = (KK * HID + T - 1) / T;

    build_x0_kernel<<<g_feat, T, 0, stream>>>(inputs, state_t, x0, X);
    prep_w_kernel<<<g_w, T, 0, stream>>>(weights, WT);

    // SpMM grid: x = 12 blocks of 256 over the 3072-wide feature axis,
    // y = node index (uniform per block -> scalar edge-list loads).
    dim3 gspmm(CB / T, NN);
    const int NNZ = NN * DEG;
    for (int s = 0; s < 2; ++s) {
        const int*   cols = sup_cols + (size_t)s * NNZ;
        const float* vals = sup_vals + (size_t)s * NNZ;
        spmm1_kernel<<<gspmm, T, 0, stream>>>(cols, vals, x0, x1, X, 1 + 2 * s);
        spmm2_kernel<<<gspmm, T, 0, stream>>>(cols, vals, x1, x0, X, 2 + 2 * s);
    }

    gemm_tanh_kernel<<<ROWS / 64, T, 0, stream>>>(X, WT, biases, out);
}
